// VQINR_352187318701
// MI455X (gfx1250) — matrix operand order repaired and checked by reading, not compiled, not run on hardware
//
#include <hip/hip_runtime.h>
#include <math.h>

// ---------------- types for CDNA5 WMMA ----------------
typedef __attribute__((ext_vector_type(16))) __bf16 v16bf;
typedef __attribute__((ext_vector_type(8)))  __bf16 v8bf;
typedef __attribute__((ext_vector_type(8)))  float  v8f;

#define B_IMG 16
#define P_PTS 16384
#define D_LAT 256
#define K_CODE 1024
#define S_STAGE 4
#define H_DIM 512
#define L_LAYERS 5
#define IN_PAD 64          // 42 padded to 64
#define TILE_M 128
#define LDSTR 520          // padded row stride in bf16 elems (1040 B, conflict-free)
#define BUFB (TILE_M * LDSTR * 2)  // 133120 bytes per activation buffer (2 bufs = 266240 <= 320KB)

// workspace layout (bytes)
#define WS_ZQ    0                         // 16*256 f32       = 16384
#define WS_LOSS  16384                     // 16 f32           = 64
#define WS_BIAS  16448                     // 5*16*512 f32     = 163840
#define WS_W0P   180288                    // 512*64 bf16      = 65536
#define WS_WHB   245824                    // 4*512*512 bf16   = 2097152

// ---------------- weight conversion ----------------
__global__ void vqinr_conv_kernel(const float* __restrict__ Wh,
                                  const float* __restrict__ W0,
                                  __bf16* __restrict__ Whb,
                                  __bf16* __restrict__ W0p) {
    int i = blockIdx.x * 256 + threadIdx.x;
    const int NWH = 4 * H_DIM * H_DIM;
    if (i < NWH) {
        Whb[i] = (__bf16)Wh[i];
    } else {
        int j = i - NWH;
        if (j < H_DIM * IN_PAD) {
            int n = j >> 6, kk = j & 63;
            W0p[j] = (kk < 42) ? (__bf16)W0[n * 42 + kk] : (__bf16)0.0f;
        }
    }
}

// ---------------- residual VQ: z_q_sum + loss partials ----------------
__global__ __launch_bounds__(256)
void vqinr_vq_kernel(const float* __restrict__ latents,
                     const float* __restrict__ vq_emb,
                     const int* __restrict__ lidx,
                     float* __restrict__ zq_out,    // [16][256]
                     float* __restrict__ loss_out)  // [16]
{
    int b = blockIdx.x;
    int t = threadIdx.x;  // 256 threads, one per dim d
    __shared__ float zc[D_LAT];
    __shared__ float resid[D_LAT];
    __shared__ float zqs[D_LAT];
    __shared__ float rmin[256];
    __shared__ int   ridx[256];
    __shared__ float rsum[256];

    int img = lidx[b];
    float loss_acc = 0.0f;   // thread 0 only
    resid[t] = 0.0f;
    zqs[t]   = 0.0f;
    __syncthreads();

    for (int s = 0; s < S_STAGE; ++s) {
        float zst = latents[((img * S_STAGE) + s) * D_LAT + t];
        resid[t] += zst;
        zc[t] = resid[t] - zqs[t];
        __syncthreads();

        // each thread scans 4 codes; ||z||^2 is constant -> argmin(e2 - 2*dot)
        float best = 3.4e38f;
        int bestk = 0;
        for (int kk = 0; kk < 4; ++kk) {
            int k = t * 4 + kk;
            const float* Er = vq_emb + ((size_t)s * K_CODE + k) * D_LAT;
            float dot = 0.0f, e2 = 0.0f;
            for (int d = 0; d < D_LAT; ++d) {
                float e = Er[d];
                dot += zc[d] * e;
                e2  += e * e;
            }
            float dist = e2 - 2.0f * dot;
            if (dist < best) { best = dist; bestk = k; }
        }
        rmin[t] = best; ridx[t] = bestk;
        __syncthreads();
        for (int off = 128; off > 0; off >>= 1) {
            if (t < off) {
                float d2 = rmin[t + off]; int i2 = ridx[t + off];
                if (d2 < rmin[t] || (d2 == rmin[t] && i2 < ridx[t])) {
                    rmin[t] = d2; ridx[t] = i2;
                }
            }
            __syncthreads();
        }
        int kbest = ridx[0];

        float zq = vq_emb[((size_t)s * K_CODE + kbest) * D_LAT + t];
        float diff = zq - zc[t];
        rsum[t] = diff * diff;
        __syncthreads();
        for (int off = 128; off > 0; off >>= 1) {
            if (t < off) rsum[t] += rsum[t + off];
            __syncthreads();
        }
        if (t == 0) loss_acc += rsum[0];
        zqs[t] += zq;
        __syncthreads();
    }
    zq_out[b * D_LAT + t] = zqs[t];
    if (t == 0) loss_out[b] = loss_acc;
}

// ---------------- fused bias table: betas + b_mod + (b0|bh), plus loss finalize ----
__global__ __launch_bounds__(256)
void vqinr_bias_kernel(const float* __restrict__ Wmod,
                       const float* __restrict__ bmod,
                       const float* __restrict__ b0,
                       const float* __restrict__ bh,
                       const float* __restrict__ zq,
                       const float* __restrict__ lossp,
                       float* __restrict__ Bias,
                       float* __restrict__ out_loss)
{
    int bx = blockIdx.x;          // = l*16 + b
    int l = bx >> 4, b = bx & 15;
    int t = threadIdx.x;
    const float* z = zq + b * D_LAT;
    for (int rep = 0; rep < 2; ++rep) {
        int n = t + rep * 256;
        const float* wr = Wmod + ((size_t)(l * H_DIM + n)) * D_LAT;
        float acc = bmod[l * H_DIM + n] + ((l == 0) ? b0[n] : bh[(l - 1) * H_DIM + n]);
        for (int d = 0; d < D_LAT; ++d) acc += z[d] * wr[d];
        Bias[(size_t)bx * H_DIM + n] = acc;
    }
    if (bx == 0 && t == 0) {
        float s = 0.0f;
        for (int i = 0; i < B_IMG; ++i) s += lossp[i];
        out_loss[0] = s * (0.25f / (4.0f * 16.0f * 256.0f));  // COMMIT / (S * B * D)
    }
}

// ---------------- WMMA fragment helpers ----------------
__device__ __forceinline__ v16bf cat8(v8bf lo, v8bf hi) {
    return __builtin_shufflevector(lo, hi, 0,1,2,3,4,5,6,7,8,9,10,11,12,13,14,15);
}

// A fragment (16x32 bf16): lane<16 rows M=lane, K={0..7,16..23}; lane>=16 K={8..15,24..31}
__device__ __forceinline__ v16bf load_a_frag(const __bf16* buf, int rowBase, int k,
                                             int l16, int hiSel) {
    const __bf16* p = buf + (rowBase + l16) * LDSTR + k + (hiSel ? 8 : 0);
    v8bf lo = *(const v8bf*)p;          // ds_load_b128
    v8bf hi = *(const v8bf*)(p + 16);   // K+16 block
    return cat8(lo, hi);
}

// B fragment (32x16 bf16): lane n=lane&15, lanes<16 hold K {0..7,16..23}, lanes>=16 hold K {8..15,24..31}
__device__ __forceinline__ v16bf load_b_frag(const __bf16* __restrict__ W, int kdim,
                                             int n, int k, int hiSel) {
    const __bf16* p = W + (size_t)n * kdim + k + (hiSel ? 8 : 0);
    v8bf lo = *(const v8bf*)p;          // global_load_b128
    v8bf hi = *(const v8bf*)(p + 16);
    return cat8(lo, hi);
}

// One dense layer: [128 x kdim] (LDS, bf16) x [H_DIM x kdim]^T (global, bf16) -> relu -> LDS bf16
// Wave layout: 2 waves over M (64 rows each), 4 waves over N (128 cols each, 2 chunks of 64).
// Per k-step: 4 A frags + 4 B frags feed 16 WMMAs (0.25 loads/WMMA on each path).
__device__ __forceinline__
void gemm_layer(const __bf16* in, __bf16* out,
                const __bf16* __restrict__ Wg, int kdim,
                const float* __restrict__ biasRow,
                int wm, int wn, int l16, int hiSel)
{
    const int rowBase = wm * 64;
    for (int nb = 0; nb < 2; ++nb) {
        const int colBase = wn * 128 + nb * 64;
        v8f acc[16];   // acc[ct*4 + rt]
#pragma unroll
        for (int i = 0; i < 16; ++i) acc[i] = (v8f)(0.0f);

        for (int k = 0; k < kdim; k += 32) {
            v16bf a[4];
#pragma unroll
            for (int rt = 0; rt < 4; ++rt)
                a[rt] = load_a_frag(in, rowBase + rt * 16, k, l16, hiSel);
#pragma unroll
            for (int ct = 0; ct < 4; ++ct) {
                v16bf bf = load_b_frag(Wg, kdim, colBase + ct * 16 + l16, k, hiSel);
#pragma unroll
                for (int rt = 0; rt < 4; ++rt) {
                    acc[ct * 4 + rt] = __builtin_amdgcn_wmma_f32_16x16x32_bf16(
                        false, a[rt], false, bf, (short)0, acc[ct * 4 + rt],
                        false, false);
                }
            }
        }
        // epilogue: + fused bias, relu, bf16 -> LDS
#pragma unroll
        for (int ct = 0; ct < 4; ++ct) {
            int n = colBase + ct * 16 + l16;
            float bias = biasRow[n];
#pragma unroll
            for (int rt = 0; rt < 4; ++rt) {
                v8f c = acc[ct * 4 + rt];
                int rb = rowBase + rt * 16 + (hiSel ? 8 : 0);
#pragma unroll
                for (int j = 0; j < 8; ++j) {
                    float f = c[j] + bias;
                    f = fmaxf(f, 0.0f);
                    out[(rb + j) * LDSTR + n] = (__bf16)f;
                }
            }
        }
    }
}

// ---------------- main MLP kernel: one 128-point tile per workgroup ----------------
__global__ __launch_bounds__(256)
void vqinr_mlp_kernel(const float* __restrict__ coords,
                      const __bf16* __restrict__ W0p,
                      const __bf16* __restrict__ Whb,
                      const float* __restrict__ Wl,
                      const float* __restrict__ bl,
                      const float* __restrict__ Bias,
                      float* __restrict__ out)
{
    extern __shared__ char smem[];
    __bf16* buf0 = (__bf16*)smem;
    __bf16* buf1 = (__bf16*)(smem + BUFB);

    const int t = threadIdx.x;
    const int b = blockIdx.x >> 7;       // 128 tiles per image
    const int tile = blockIdx.x & 127;
    const int pBase = tile * TILE_M;

    // ---- stage positional encoding into buf0 (bf16, padded to 64) ----
    if (t < TILE_M) {
        int p = pBase + t;
        float x = coords[((size_t)b * P_PTS + p) * 2 + 0];
        float y = coords[((size_t)b * P_PTS + p) * 2 + 1];
        __bf16* row = buf0 + t * LDSTR;
        row[0] = (__bf16)x;
        row[1] = (__bf16)y;
        float fr = 3.14159265358979323846f;
        for (int f = 0; f < 10; ++f) {
            float ax = x * fr, ay = y * fr;
            row[2 + f]  = (__bf16)sinf(ax);
            row[12 + f] = (__bf16)sinf(ay);
            row[22 + f] = (__bf16)cosf(ax);
            row[32 + f] = (__bf16)cosf(ay);
            fr *= 2.0f;
        }
        for (int i = 42; i < IN_PAD; ++i) row[i] = (__bf16)0.0f;
    }
    __syncthreads();

    const int wave = t >> 5, lane = t & 31;
    const int wm = wave & 1, wn = wave >> 1;
    const int l16 = lane & 15, hiSel = lane >> 4;

    // layer 0: K=64
    gemm_layer(buf0, buf1, W0p, IN_PAD, Bias + (size_t)(0 * 16 + b) * H_DIM,
               wm, wn, l16, hiSel);
    __syncthreads();

    __bf16* bin = buf1; __bf16* bout = buf0;
    for (int l = 0; l < 4; ++l) {
        gemm_layer(bin, bout, Whb + (size_t)l * H_DIM * H_DIM, H_DIM,
                   Bias + (size_t)((l + 1) * 16 + b) * H_DIM, wm, wn, l16, hiSel);
        __syncthreads();
        __bf16* tmp = bin; bin = bout; bout = tmp;
    }

    // ---- output head: 512 -> 3, VALU; each thread handles one row, all 3 outputs ----
    if (t < TILE_M) {
        int row = t;
        const __bf16* hr = bin + row * LDSTR;
        float a0 = bl[0], a1 = bl[1], a2 = bl[2];
        const float* wl0 = Wl;
        const float* wl1 = Wl + H_DIM;
        const float* wl2 = Wl + 2 * H_DIM;
        for (int k = 0; k < H_DIM; ++k) {
            float hv = (float)hr[k];
            a0 += hv * wl0[k];
            a1 += hv * wl1[k];
            a2 += hv * wl2[k];
        }
        size_t o = ((size_t)b * P_PTS + pBase + row) * 3;
        out[o + 0] = a0;
        out[o + 1] = a1;
        out[o + 2] = a2;
    }
}

// ---------------- launcher ----------------
extern "C" void kernel_launch(void* const* d_in, const int* in_sizes, int n_in,
                              void* d_out, int out_size, void* d_ws, size_t ws_size,
                              hipStream_t stream) {
    const float* coords  = (const float*)d_in[0];
    const float* latents = (const float*)d_in[1];
    const float* vq_emb  = (const float*)d_in[2];
    const float* W_mod   = (const float*)d_in[3];
    const float* b_mod   = (const float*)d_in[4];
    const float* W0      = (const float*)d_in[5];
    const float* b0      = (const float*)d_in[6];
    const float* Wh      = (const float*)d_in[7];
    const float* bh      = (const float*)d_in[8];
    const float* Wl      = (const float*)d_in[9];
    const float* bl      = (const float*)d_in[10];
    const int*   lidx    = (const int*)d_in[11];

    float* out = (float*)d_out;
    char*  ws  = (char*)d_ws;
    float*  ws_zq   = (float*)(ws + WS_ZQ);
    float*  ws_loss = (float*)(ws + WS_LOSS);
    float*  ws_bias = (float*)(ws + WS_BIAS);
    __bf16* ws_w0p  = (__bf16*)(ws + WS_W0P);
    __bf16* ws_whb  = (__bf16*)(ws + WS_WHB);

    // 1. weight conversion (f32 -> bf16, pad W0 to 64)
    vqinr_conv_kernel<<<4224, 256, 0, stream>>>(Wh, W0, ws_whb, ws_w0p);
    // 2. residual VQ per image
    vqinr_vq_kernel<<<B_IMG, 256, 0, stream>>>(latents, vq_emb, lidx, ws_zq, ws_loss);
    // 3. fused bias table + vq_loss scalar
    vqinr_bias_kernel<<<L_LAYERS * B_IMG, 256, 0, stream>>>(
        W_mod, b_mod, b0, bh, ws_zq, ws_loss, ws_bias,
        out + (size_t)B_IMG * P_PTS * 3);
    // 4. WMMA MLP over all points (128-pt tiles, 266KB LDS ping-pong)
    vqinr_mlp_kernel<<<B_IMG * (P_PTS / TILE_M), 256, 2 * BUFB, stream>>>(
        coords, ws_w0p, ws_whb, Wl, bl, ws_bias, out);
}
